// FRUCell_50809463112403
// MI455X (gfx1250) — matrix operand order repaired and checked by reading, not compiled, not run on hardware
//
#include <hip/hip_runtime.h>

// ---------------------------------------------------------------------------
// FRU cell for gfx1250 (MI455X), wave32 + WMMA bf16 + async global->LDS.
//
// Phase 1 (prep): bf16 transposed weights + cos table.
// Phase 2 (scan): 1 workgroup, 32 waves, sequential over T; recur/stats GEMMs
//                 via v_wmma_f32_16x16x32_bf16; stores bf16 states to ws.
// Phase 3 (gemm): outs = gelu(states @ Wout + b): 128x128 tiles, double-
//                 buffered LDS staged with global_load_async_to_lds_b128.
// ---------------------------------------------------------------------------

typedef __attribute__((ext_vector_type(16))) __bf16 v16bf;
typedef __attribute__((ext_vector_type(8)))  float  v8f;

#define BB 32
#define TT 2048
#define DD 128
#define FF 16
#define SS 64
#define RR 64
#define STATE 1024
#define CATK 192   // D + R

__device__ __forceinline__ unsigned short f2bf(float f) {
    union { float f; unsigned int u; } v; v.f = f;
    unsigned int r = (v.u + 0x7FFFu + ((v.u >> 16) & 1u)) >> 16;
    return (unsigned short)r;
}

__device__ __forceinline__ float gelu_exact(float v) {
    return 0.5f * v * (1.0f + erff(v * 0.70710678118654752f));
}

__device__ __forceinline__ v8f wmma_bf16(v16bf a, v16bf b, v8f c) {
    return __builtin_amdgcn_wmma_f32_16x16x32_bf16(
        /*neg_a=*/false, a, /*neg_b=*/false, b,
        /*c_mod=*/(short)0, c, /*reuse_a=*/false, /*reuse_b=*/false);
}

// A fragment (16x32 bf16, MxK): lane half: VGPR0-3 = K[half*8 .. +7],
// VGPR4-7 = K[16+half*8 .. +7]. Two 16B loads.
__device__ __forceinline__ v16bf fragA(const unsigned short* base, int row,
                                       int stride, int k0, int half) {
    union { uint4 u[2]; v16bf v; } t;
    const unsigned short* p = base + row * stride + k0;
    t.u[0] = *(const uint4*)(p + half * 8);
    t.u[1] = *(const uint4*)(p + 16 + half * 8);
    return t.v;
}

// B fragment (32x16 bf16, KxN) loaded from B^T[n][k]: lane half: VGPR0-3 =
// K[half*8 .. +7], VGPR4-7 = K[16+half*8 .. +7]. Two 16B loads.
__device__ __forceinline__ v16bf fragB(const unsigned short* baseT, int n,
                                       int stride, int k0, int half) {
    union { uint4 u[2]; v16bf v; } t;
    const unsigned short* p = baseT + n * stride + k0;
    t.u[0] = *(const uint4*)(p + half * 8);
    t.u[1] = *(const uint4*)(p + 16 + half * 8);
    return t.v;
}

// Async DMA: 16 bytes per lane, global -> LDS, tracked by ASYNCcnt.
__device__ __forceinline__ void async_b128(unsigned lds_byte_addr,
                                           const void* sbase, unsigned voff) {
    asm volatile("global_load_async_to_lds_b128 %0, %1, %2"
                 :: "v"(lds_byte_addr), "v"(voff), "s"(sbase)
                 : "memory");
}

__device__ __forceinline__ void wait_asynccnt0() {
    asm volatile("s_wait_asynccnt 0" ::: "memory");
}

// ---------------------------------------------------------------------------
// Phase 1: weight transposes -> bf16, cos table with 1/L folded in.
// ---------------------------------------------------------------------------
#define NWOUT (1024 * 1024)
#define NWR   (64 * 1024)
#define NWS   (64 * 192)
#define NCOS  (2048 * 16)

__global__ void fru_prep(const float* __restrict__ Wout,
                         const float* __restrict__ Wr,
                         const float* __restrict__ Ws,
                         const float* __restrict__ phases,
                         unsigned short* __restrict__ WoutT,
                         unsigned short* __restrict__ WrT,
                         unsigned short* __restrict__ WsT,
                         float* __restrict__ costab) {
    int i = blockIdx.x * blockDim.x + threadIdx.x;
    if (i < NWOUT) {
        int n = i >> 10, k = i & 1023;
        WoutT[i] = f2bf(Wout[k * 1024 + n]);
    } else if (i < NWOUT + NWR) {
        int j = i - NWOUT;
        int n = j >> 10, k = j & 1023;
        WrT[j] = f2bf(Wr[k * 64 + n]);
    } else if (i < NWOUT + NWR + NWS) {
        int j = i - NWOUT - NWR;
        int n = j / 192, k = j % 192;
        WsT[j] = f2bf(Ws[k * 64 + n]);
    } else if (i < NWOUT + NWR + NWS + NCOS) {
        int j = i - NWOUT - NWR - NWS;
        int t = j >> 4, f = j & 15;
        const float TWO_PI = 6.28318530717958648f;
        float ang = (TWO_PI / 2048.0f) * (float)(t + 1) * (float)f +
                    TWO_PI * phases[f];
        costab[j] = (1.0f / 2048.0f) * cosf(ang);
    }
}

// ---------------------------------------------------------------------------
// Phase 2: sequential scan. One workgroup, 1024 threads (32 waves).
//   wave w = batch row b (32 f32 state regs/thread, chunk e = lane*32+j,
//   so frequency f = lane>>1 is fixed per thread).
//   recur GEMM: 32 waves = (2 Mtile) x (4 Ntile) x (4-way K split), LDS reduce.
//   stats GEMM: waves 0..7, full K=192.
// Dynamic LDS: stateBF 64K | catBF 12K | partial 32K | statsF 8K | cos 64B
// ---------------------------------------------------------------------------
__global__ __launch_bounds__(1024) void fru_scan(
    const float* __restrict__ x,              // (32,2048,128)
    const unsigned short* __restrict__ WrT,   // (64,1024)
    const float* __restrict__ br,             // (64)
    const unsigned short* __restrict__ WsT,   // (64,192)
    const float* __restrict__ bs,             // (64)
    const float* __restrict__ costab,         // (2048,16)
    unsigned short* __restrict__ statesBF)    // (2048,32,1024) = (t*32+b, e)
{
    extern __shared__ char smem[];
    unsigned short* stateBF = (unsigned short*)smem;               // 32*1024
    unsigned short* catBF   = stateBF + 32 * 1024;                 // 32*192
    float* partial = (float*)(catBF + 32 * 192);                   // 4*8*256
    float* statsF  = partial + 8192;                               // 32*64
    float* ldsCos  = statsF + 2048;                                // 16

    const int tid  = threadIdx.x;
    const int wave = tid >> 5;
    const int lane = tid & 31;
    const int half = lane >> 4;
    const int mq   = lane & 15;
    const int b    = wave;                 // batch row owned for update

    // recur GEMM assignment: wave = kg*8 + mt*4 + nt
    const int mt = (wave >> 2) & 1;
    const int nt = wave & 3;
    const int kg = wave >> 3;

    // update-phase constants: e = lane*32 + j  ->  f = lane>>1, s = (lane&1)*32+j
    const int fidx  = lane >> 1;
    const int sbase = (lane & 1) * 32;
    const float fm  = (fidx == 0) ? 0.0f : 1.0f;

    float st[32];
#pragma unroll
    for (int j = 0; j < 32; ++j) st[j] = 0.0f;

    {   // zero bf16 state mirror
        uint4 z = make_uint4(0u, 0u, 0u, 0u);
        uint4* p = (uint4*)(stateBF + b * 1024 + lane * 32);
        p[0] = z; p[1] = z; p[2] = z; p[3] = z;
    }
    __syncthreads();

    for (int t = 0; t < TT; ++t) {
        // ---- stage x_t into cat (cols 0..127) + cos row --------------------
        {
            const float4 xv = *(const float4*)(x + ((size_t)b * TT + t) * DD + lane * 4);
            unsigned short* cp = catBF + b * CATK + lane * 4;
            cp[0] = f2bf(xv.x); cp[1] = f2bf(xv.y);
            cp[2] = f2bf(xv.z); cp[3] = f2bf(xv.w);
            if (tid < 16) ldsCos[tid] = costab[t * 16 + tid];
            if (t + 1 < TT)   // hide next step's x latency (global_prefetch_b8)
                __builtin_prefetch(x + ((size_t)b * TT + (t + 1)) * DD + lane * 4, 0, 3);
        }

        // ---- recur = state @ WrT^T : M=32 N=64 K=1024, 4-way K split -------
        v8f acc = {0.f, 0.f, 0.f, 0.f, 0.f, 0.f, 0.f, 0.f};
#pragma unroll 2
        for (int kc = 0; kc < 8; ++kc) {
            const int k0 = (kg * 8 + kc) * 32;
            v16bf a  = fragA(stateBF, mt * 16 + mq, STATE, k0, half);
            v16bf bf = fragB(WrT,     nt * 16 + mq, STATE, k0, half);
            acc = wmma_bf16(a, bf, acc);
        }
        {
            float* pp = partial + (kg * 8 + (mt * 4 + nt)) * 256;
#pragma unroll
            for (int v = 0; v < 8; ++v) pp[v * 32 + lane] = acc[v];
        }
        __syncthreads();

        // ---- reduce K-split partials, bias + gelu -> cat cols 128..191 -----
#pragma unroll
        for (int r = 0; r < 2; ++r) {
            const int p    = tid + r * 1024;        // 0..2047
            const int tile = p >> 8;
            const int idx  = p & 255;
            float s = partial[tile * 256 + idx] + partial[(8 + tile) * 256 + idx] +
                      partial[(16 + tile) * 256 + idx] + partial[(24 + tile) * 256 + idx];
            const int v  = idx >> 5;
            const int l  = idx & 31;
            const int mm = ((tile >> 2) & 1) * 16 + v + 8 * (l >> 4);
            const int nn = (tile & 3) * 16 + (l & 15);
            catBF[mm * CATK + DD + nn] = f2bf(gelu_exact(s + br[nn]));
        }
        __syncthreads();

        // ---- stats = gelu([x,recur] @ WsT^T) : M=32 N=64 K=192, waves 0..7 -
        if (wave < 8) {
            v8f acc2 = {0.f, 0.f, 0.f, 0.f, 0.f, 0.f, 0.f, 0.f};
#pragma unroll
            for (int kc = 0; kc < 6; ++kc) {
                const int k0 = kc * 32;
                v16bf a  = fragA(catBF, mt * 16 + mq, CATK, k0, half);
                v16bf bf = fragB(WsT,   nt * 16 + mq, CATK, k0, half);
                acc2 = wmma_bf16(a, bf, acc2);
            }
#pragma unroll
            for (int v = 0; v < 8; ++v) {
                const int mm = mt * 16 + v + 8 * half;
                const int nn = nt * 16 + mq;
                statsF[mm * SS + nn] = gelu_exact(acc2[v] + bs[nn]);
            }
        }
        __syncthreads();

        // ---- state update + bf16 mirror + global bf16 store ----------------
        {
            const float c = ldsCos[fidx];
            const float* sp = statsF + b * SS + sbase;
            unsigned short obuf[32];
#pragma unroll
            for (int j = 0; j < 32; ++j) {
                st[j] = fm * st[j] + c * sp[j];
                obuf[j] = f2bf(st[j]);
            }
            uint4* lv = (uint4*)(stateBF + b * 1024 + lane * 32);
            uint4* gv = (uint4*)(statesBF + ((size_t)t * 32 + b) * 1024 + lane * 32);
            const uint4* ov = (const uint4*)obuf;
#pragma unroll
            for (int q = 0; q < 4; ++q) { lv[q] = ov[q]; gv[q] = ov[q]; }
        }
        __syncthreads();
    }
}

// ---------------------------------------------------------------------------
// Phase 3: outs = gelu(states_bf16 @ Wout + bias).  M=65536, N=1024, K=1024.
// 256 threads (8 waves), 128x128 tile, K-chunks of 32. Double-buffered LDS
// staged by global_load_async_to_lds_b128 (ASYNCcnt), one barrier per chunk.
// Wave w owns M-subtile w (16 rows) x all 8 N-subtiles -> 8 WMMA per chunk.
// Pipeline: issue(k+1) | compute(k) | s_wait_asynccnt 0 | barrier.
// ---------------------------------------------------------------------------
__global__ __launch_bounds__(256) void fru_out_gemm(
    const unsigned short* __restrict__ A,     // (65536,1024) bf16
    const unsigned short* __restrict__ Bt,    // (1024,1024) bf16 = Wout^T [n][k]
    const float* __restrict__ bias,           // (1024)
    float* __restrict__ out)                  // (32,2048,1024) f32
{
    __shared__ unsigned short ldsA[2][128 * 32];   // 2 x 8 KB
    __shared__ unsigned short ldsB[2][128 * 32];   // 2 x 8 KB

    const int tid  = threadIdx.x;
    const int wave = tid >> 5;
    const int lane = tid & 31;
    const int half = lane >> 4;
    const int mq   = lane & 15;

    const int n0 = blockIdx.x * 128;
    const int r0 = blockIdx.y * 128;

    // staging: each thread moves 32 B of A and 32 B of B per chunk
    const int srow = tid >> 1;             // 0..127
    const int scol = (tid & 1) * 16;       // 0 or 16 (elements)

    const unsigned ldsAaddr[2] = {
        (unsigned)(uintptr_t)&ldsA[0][srow * 32 + scol],
        (unsigned)(uintptr_t)&ldsA[1][srow * 32 + scol]};
    const unsigned ldsBaddr[2] = {
        (unsigned)(uintptr_t)&ldsB[0][srow * 32 + scol],
        (unsigned)(uintptr_t)&ldsB[1][srow * 32 + scol]};

    const unsigned agoff = ((unsigned)(r0 + srow) * 1024u + (unsigned)scol) * 2u;
    const unsigned bgoff = ((unsigned)(n0 + srow) * 1024u + (unsigned)scol) * 2u;

    auto issue = [&](int kc, int buf) {
        const unsigned kb = (unsigned)(kc * 32) * 2u;  // K byte offset
        async_b128(ldsAaddr[buf],      A,  agoff + kb);
        async_b128(ldsAaddr[buf] + 16, A,  agoff + kb + 16);
        async_b128(ldsBaddr[buf],      Bt, bgoff + kb);
        async_b128(ldsBaddr[buf] + 16, Bt, bgoff + kb + 16);
    };

    v8f acc[8];
    {
        const v8f z = {0.f, 0.f, 0.f, 0.f, 0.f, 0.f, 0.f, 0.f};
#pragma unroll
        for (int i = 0; i < 8; ++i) acc[i] = z;
    }

    issue(0, 0);
    wait_asynccnt0();
    __syncthreads();

    for (int kc = 0; kc < 32; ++kc) {
        const int buf = kc & 1;
        if (kc + 1 < 32) issue(kc + 1, buf ^ 1);   // DMA overlaps WMMA below

        v16bf a = fragA(ldsA[buf], wave * 16 + mq, 32, 0, half);
#pragma unroll
        for (int ntc = 0; ntc < 8; ++ntc) {
            v16bf bf = fragB(ldsB[buf], ntc * 16 + mq, 32, 0, half);
            acc[ntc] = wmma_bf16(a, bf, acc[ntc]);
        }

        wait_asynccnt0();      // own chunk kc+1 writes landed
        __syncthreads();       // everyone's writes visible, buf[kc] reads done
    }

#pragma unroll
    for (int ntc = 0; ntc < 8; ++ntc) {
#pragma unroll
        for (int v = 0; v < 8; ++v) {
            const int mm = wave * 16 + v + 8 * half;
            const int r  = r0 + mm;
            const int bb = r & 31;
            const int t  = r >> 5;
            const int nn = n0 + ntc * 16 + mq;
            out[((size_t)bb * TT + t) * 1024 + nn] = gelu_exact(acc[ntc][v] + bias[nn]);
        }
    }
}

// ---------------------------------------------------------------------------
// Host launcher. ws layout (bytes):
//   [0, 128MB)  states bf16 | +2MB WoutT | +128KB WrT | +24KB WsT | +128KB cos
// ---------------------------------------------------------------------------
extern "C" void kernel_launch(void* const* d_in, const int* in_sizes, int n_in,
                              void* d_out, int out_size, void* d_ws, size_t ws_size,
                              hipStream_t stream) {
    const float* x      = (const float*)d_in[0];
    const float* Wr     = (const float*)d_in[1];
    const float* br     = (const float*)d_in[2];
    const float* Ws     = (const float*)d_in[3];
    const float* bs     = (const float*)d_in[4];
    const float* Wout   = (const float*)d_in[5];
    const float* bo     = (const float*)d_in[6];
    const float* phases = (const float*)d_in[7];
    float* out = (float*)d_out;

    char* ws = (char*)d_ws;
    const size_t OFF_WOUTT  = 134217728ull;            // 128 MB
    const size_t OFF_WRT    = OFF_WOUTT + 2097152ull;  // +2 MB
    const size_t OFF_WST    = OFF_WRT + 131072ull;     // +128 KB
    const size_t OFF_COS    = OFF_WST + 24576ull;      // +24 KB

    unsigned short* statesBF = (unsigned short*)(ws);
    unsigned short* WoutT    = (unsigned short*)(ws + OFF_WOUTT);
    unsigned short* WrT      = (unsigned short*)(ws + OFF_WRT);
    unsigned short* WsT      = (unsigned short*)(ws + OFF_WST);
    float*          costab   = (float*)(ws + OFF_COS);

    // Phase 1: prep
    const int prep_total = NWOUT + NWR + NWS + NCOS;
    fru_prep<<<(prep_total + 255) / 256, 256, 0, stream>>>(
        Wout, Wr, Ws, phases, WoutT, WrT, WsT, costab);

    // Phase 2: sequential scan, one WGP-resident workgroup
    const size_t shbytes = 65536 + 12288 + 32768 + 8192 + 64;
    fru_scan<<<1, 1024, shbytes, stream>>>(x, WrT, br, WsT, bs, costab, statesBF);

    // Phase 3: big WMMA GEMM + gelu, async double-buffered staging
    fru_out_gemm<<<dim3(8, 512), 256, 0, stream>>>(statesBF, WoutT, bo, out);
}
